// Attention_global_79413945303289
// MI455X (gfx1250) — matrix operand order repaired and checked by reading, not compiled, not run on hardware
//
#include <hip/hip_runtime.h>
#include <math.h>
#include <stdint.h>

typedef __attribute__((ext_vector_type(16))) _Float16 v16h;
typedef __attribute__((ext_vector_type(8)))  _Float16 v8h;
typedef __attribute__((ext_vector_type(8)))  float    v8f;
typedef __attribute__((ext_vector_type(4)))  float    v4f;

#define C_DIM 256
#define S_DIM 1024
#define BK    32              // keys per block iteration
#define QTILE 128             // queries per workgroup (8 waves * 16)
#define NWAVE 8
#define CS_STRIDE 40          // Kcs [c][s] row stride in halves (BK + 8 pad) = 80B (16B aligned)
#define SC_STRIDE 264         // Ksc [s][c] row stride in halves (C + 8 pad) = 528B (16B aligned)
#define P_STRIDE  40          // P staging [16][BK] row stride in halves

// 16B async DMA copy: global (saddr base + 32-bit voffset) -> LDS byte address.
// GLOBAL_LOAD_ASYNC_TO_LDS_B128, tracked by ASYNCcnt (CDNA5 ISA).
__device__ __forceinline__ void async_copy_b128(uint32_t lds_addr, const void* gbase,
                                                uint32_t goff) {
    asm volatile("global_load_async_to_lds_b128 %0, %1, %2"
                 :: "v"(lds_addr), "v"(goff), "s"(gbase)
                 : "memory");
}

// ---- one-shot prep: K (x_global) f32 -> f16 in both orientations in workspace ----
__global__ void prep_k_f16(const float* __restrict__ x_glob,
                           _Float16* __restrict__ kcs,   // [C][S]  (identity layout)
                           _Float16* __restrict__ ksc) { // [S][C]
    int e = blockIdx.x * blockDim.x + threadIdx.x;   // 0 .. C*S-1
    int c = e >> 10;          // / S_DIM
    int s = e & (S_DIM - 1);
    _Float16 h = (_Float16)x_glob[e];
    kcs[e] = h;
    ksc[s * C_DIM + c] = h;
}

template <bool USE_ASYNC>
__global__ __launch_bounds__(256, 1)
void attn_global_fa_kernel(const float* __restrict__ x_fpn,
                           const float* __restrict__ x_glob,
                           const _Float16* __restrict__ kcs_g,
                           const _Float16* __restrict__ ksc_g,
                           float* __restrict__ out)
{
    __shared__ __attribute__((aligned(16))) _Float16 Kcs[C_DIM * CS_STRIDE];     // 20 KB
    __shared__ __attribute__((aligned(16))) _Float16 Ksc[BK * SC_STRIDE];        // 16.5 KB
    __shared__ __attribute__((aligned(16))) _Float16 Pst[NWAVE * 16 * P_STRIDE]; // 10 KB

    const int tid  = threadIdx.x;
    const int w    = tid >> 5;
    const int lane = tid & 31;
    const int col  = lane & 15;
    const int hi   = (lane >> 4) & 1;   // which 16-lane half of the wave32

    const int n     = blockIdx.x >> 3;       // batch
    const int qt    = blockIdx.x & 7;        // query tile in batch
    const int qbase = qt * QTILE + w * 16;   // first query row of this wave

    const uint32_t kcs_lds = (uint32_t)(uintptr_t)(&Kcs[0]);
    const uint32_t ksc_lds = (uint32_t)(uintptr_t)(&Ksc[0]);

    // ---- Load Q A-fragments once: Q[q][c] = x_fpn[n][q*256 + c] (raw reshape) ----
    // A 16-bit 16x32 layout: M = lane%16 ; K = h + (h>=8?8:0) + (lane>=16?8:0)
    const float* qrow = x_fpn + (size_t)n * (C_DIM * S_DIM) + (size_t)(qbase + col) * C_DIM;
    v16h qfrag[8];
#pragma unroll
    for (int kc = 0; kc < 8; ++kc) {
        const int c0 = kc * 32 + hi * 8;
        v4f f0 = *(const v4f*)(qrow + c0);
        v4f f1 = *(const v4f*)(qrow + c0 + 4);
        v4f f2 = *(const v4f*)(qrow + c0 + 16);
        v4f f3 = *(const v4f*)(qrow + c0 + 20);
#pragma unroll
        for (int i = 0; i < 4; ++i) {
            qfrag[kc][i]      = (_Float16)f0[i];
            qfrag[kc][i + 4]  = (_Float16)f1[i];
            qfrag[kc][i + 8]  = (_Float16)f2[i];
            qfrag[kc][i + 12] = (_Float16)f3[i];
        }
    }

    // ---- Flash-attention state ----
    v8f o_acc[16];
    const v8f vzero = {0.f, 0.f, 0.f, 0.f, 0.f, 0.f, 0.f, 0.f};
#pragma unroll
    for (int ct = 0; ct < 16; ++ct) o_acc[ct] = vzero;
    float m_run[8], l_run[8];
#pragma unroll
    for (int r = 0; r < 8; ++r) { m_run[r] = -INFINITY; l_run[r] = 0.0f; }

    for (int blk = 0; blk < S_DIM / BK; ++blk) {
        __syncthreads();   // all waves finished reading the previous tile
        const int s0 = blk * BK;

        if constexpr (USE_ASYNC) {
            // ---- Async DMA staging from pre-converted f16 K (L2-resident, 1MB) ----
            // Kcs tile: 256 rows x 32 halves -> 4 x 16B chunks per row, 1024 chunks
#pragma unroll
            for (int it = 0; it < 4; ++it) {
                int idx  = it * 256 + tid;
                int c    = idx >> 2;
                int part = idx & 3;
                async_copy_b128(kcs_lds + (uint32_t)(c * CS_STRIDE + part * 8) * 2,
                                kcs_g,
                                (uint32_t)((c * S_DIM + s0 + part * 8) * 2));
            }
            // Ksc tile: 32 rows x 256 halves -> 32 x 16B chunks per row, 1024 chunks
#pragma unroll
            for (int it = 0; it < 4; ++it) {
                int idx  = it * 256 + tid;
                int sl   = idx >> 5;
                int part = idx & 31;
                async_copy_b128(ksc_lds + (uint32_t)(sl * SC_STRIDE + part * 8) * 2,
                                ksc_g,
                                (uint32_t)(((s0 + sl) * C_DIM + part * 8) * 2));
            }
            asm volatile("s_wait_asynccnt 0" ::: "memory");
        } else {
            // ---- Fallback: stage f32 K with in-flight convert ----
#pragma unroll 4
            for (int it = 0; it < (C_DIM * BK) / 256; ++it) {
                int idx = it * 256 + tid;
                int c  = idx >> 5;       // / BK
                int sl = idx & (BK - 1);
                _Float16 h16 = (_Float16)x_glob[(size_t)c * S_DIM + s0 + sl];
                Kcs[c * CS_STRIDE + sl] = h16;
                Ksc[sl * SC_STRIDE + c] = h16;
            }
            if (blk + 1 < S_DIM / BK) {   // prefetch next tile into cache hierarchy
                int c  = tid >> 5;
                int sl = tid & (BK - 1);
                __builtin_prefetch(x_glob + (size_t)c * S_DIM + (s0 + BK) + sl, 0, 1);
            }
        }
        __syncthreads();

        // ---- GEMM1: S_tile(16 x BK) = Q(16x256) @ K(256 x BK) ----
        // B 16-bit 32x16 layout: N = lane%16 ; K = h + (h>=8?8:0) + (lane>=16?8:0)
        v8f s_acc[BK / 16];
#pragma unroll
        for (int nt = 0; nt < BK / 16; ++nt) {
            v8f acc = vzero;
            const _Float16* bbase = Ksc + (nt * 16 + col) * SC_STRIDE + hi * 8;
#pragma unroll
            for (int kc = 0; kc < 8; ++kc) {
                v8h lo = *(const v8h*)(bbase + kc * 32);
                v8h hv = *(const v8h*)(bbase + kc * 32 + 16);
                v16h bf;
#pragma unroll
                for (int i = 0; i < 8; ++i) { bf[i] = lo[i]; bf[i + 8] = hv[i]; }
                acc = __builtin_amdgcn_wmma_f32_16x16x32_f16(
                        false, qfrag[kc], false, bf, (short)0, acc, false, false);
            }
            s_acc[nt] = acc;
        }

        // ---- Online softmax (C/D layout: reg r -> row r + 8*hi, lanes 0..15 = cols) ----
#pragma unroll
        for (int r = 0; r < 8; ++r) {
            float mx = s_acc[0][r];
#pragma unroll
            for (int nt = 1; nt < BK / 16; ++nt) mx = fmaxf(mx, s_acc[nt][r]);
#pragma unroll
            for (int off = 1; off <= 8; off <<= 1)
                mx = fmaxf(mx, __shfl_xor(mx, off));
            const float mnew = fmaxf(m_run[r], mx);
            const float corr = __expf(m_run[r] - mnew);
            m_run[r] = mnew;
            float rs = 0.0f;
#pragma unroll
            for (int nt = 0; nt < BK / 16; ++nt) {
                float p = __expf(s_acc[nt][r] - mnew);
                s_acc[nt][r] = p;
                rs += p;
            }
#pragma unroll
            for (int off = 1; off <= 8; off <<= 1)
                rs += __shfl_xor(rs, off);
            l_run[r] = l_run[r] * corr + rs;
#pragma unroll
            for (int ct = 0; ct < 16; ++ct)
                o_acc[ct][r] *= corr;
            // stage P (f16) row-major [16][BK] for A-fragment reload
            const int row = r + hi * 8;
#pragma unroll
            for (int nt = 0; nt < BK / 16; ++nt)
                Pst[w * 16 * P_STRIDE + row * P_STRIDE + nt * 16 + col] =
                    (_Float16)s_acc[nt][r];
        }

        // Per-wave LDS RAW: CDNA5 split counter wait (wave-local, no full barrier)
        asm volatile("s_wait_dscnt 0" ::: "memory");

        // ---- Reload P as A fragment (16 x 32) ----
        v16h pfrag;
        {
            const _Float16* pbase = Pst + w * 16 * P_STRIDE + col * P_STRIDE + hi * 8;
            v8h lo = *(const v8h*)(pbase);
            v8h hv = *(const v8h*)(pbase + 16);
#pragma unroll
            for (int i = 0; i < 8; ++i) { pfrag[i] = lo[i]; pfrag[i + 8] = hv[i]; }
        }

        // ---- GEMM2: O(16x256) += P(16x32) @ V(32x256), V[s][c] = Kcs[c][s] ----
#pragma unroll
        for (int ct = 0; ct < 16; ++ct) {
            const _Float16* vb = Kcs + (ct * 16 + col) * CS_STRIDE + hi * 8;
            v8h lo = *(const v8h*)(vb);
            v8h hv = *(const v8h*)(vb + 16);
            v16h bf;
#pragma unroll
            for (int i = 0; i < 8; ++i) { bf[i] = lo[i]; bf[i + 8] = hv[i]; }
            o_acc[ct] = __builtin_amdgcn_wmma_f32_16x16x32_f16(
                    false, pfrag, false, bf, (short)0, o_acc[ct], false, false);
        }
    }

    // ---- Normalize and write out[n][c][q] (fp32) ----
    float* obase = out + (size_t)n * (C_DIM * S_DIM);
#pragma unroll
    for (int r = 0; r < 8; ++r) {
        const float inv = 1.0f / l_run[r];
        const int q = qbase + r + hi * 8;
#pragma unroll
        for (int ct = 0; ct < 16; ++ct) {
            const int c = ct * 16 + col;
            obase[(size_t)c * S_DIM + q] = o_acc[ct][r] * inv;
        }
    }
}

extern "C" void kernel_launch(void* const* d_in, const int* in_sizes, int n_in,
                              void* d_out, int out_size, void* d_ws, size_t ws_size,
                              hipStream_t stream) {
    (void)n_in; (void)out_size;
    const float* x_fpn  = (const float*)d_in[0];
    const float* x_glob = (const float*)d_in[1];
    float* out = (float*)d_out;
    const int N = in_sizes[0] / (C_DIM * S_DIM);          // 32
    dim3 grid(N * (S_DIM / QTILE));                        // 32 * 8 = 256 workgroups

    const size_t need = (size_t)2 * C_DIM * S_DIM * sizeof(_Float16);  // 1 MB
    if (ws_size >= need && d_ws != nullptr) {
        _Float16* kcs_g = (_Float16*)d_ws;
        _Float16* ksc_g = kcs_g + (size_t)C_DIM * S_DIM;
        prep_k_f16<<<dim3((C_DIM * S_DIM) / 256), 256, 0, stream>>>(x_glob, kcs_g, ksc_g);
        attn_global_fa_kernel<true><<<grid, 256, 0, stream>>>(x_fpn, x_glob, kcs_g, ksc_g, out);
    } else {
        attn_global_fa_kernel<false><<<grid, 256, 0, stream>>>(x_fpn, x_glob, nullptr, nullptr, out);
    }
}
